// GATLayer_57080115364429
// MI455X (gfx1250) — compile-verified, matrix operand order checked by reading, not run on hardware
//
#include <hip/hip_runtime.h>

typedef __attribute__((ext_vector_type(16))) _Float16 v16h;
typedef __attribute__((ext_vector_type(8)))  float    v8f;

#define NN   100000
#define NE   1600000
#define INF_ 128
#define NH   8
#define NF   16
#define HF   128
#define NEG_SLOPE 0.2f
#define EPSV 1e-16f

__device__ __forceinline__ void atomicMaxF(float* addr, float val) {
  if (val >= 0.0f) atomicMax((int*)addr, __float_as_int(val));
  else             atomicMin((unsigned int*)addr, __float_as_uint(val));
}

__global__ void gat_init_kernel(float* __restrict__ denom, float* __restrict__ smax) {
  int i = blockIdx.x * blockDim.x + threadIdx.x;
  if (i < NN * NH) denom[i] = 0.0f;
  if (i == 0) *(unsigned int*)smax = 0xFF800000u;  // -inf
}

// One block = one 16-row node tile. 8 waves; wave w computes output cols [16w,16w+16)
// of both projections (== head w of p), via v_wmma_f32_16x16x32_f16, and derives
// attention scores for head w with an in-register lane reduction.
__global__ __launch_bounds__(256)
void gat_gemm_scores_kernel(const float* __restrict__ x,
                            const float* __restrict__ Wp,
                            const float* __restrict__ Ws,
                            const float* __restrict__ a_src,
                            const float* __restrict__ a_tgt,
                            float* __restrict__ p,
                            float* __restrict__ ssrc,
                            float* __restrict__ stgt,
                            float* __restrict__ out) {
  __shared__ _Float16 xt[16][INF_];
  const int row0 = blockIdx.x * 16;
  const int tid  = threadIdx.x;

  for (int idx = tid; idx < 16 * INF_; idx += 256) {
    int r = idx >> 7, c = idx & 127;
    xt[r][c] = (_Float16)x[(size_t)(row0 + r) * INF_ + c];
  }
  __syncthreads();

  const int wid  = tid >> 5;        // wave id == n-tile == head
  const int lane = tid & 31;
  const int id   = lane & 15;       // N / M index within tile
  const int h2   = lane >> 4;       // lane half

  v8f accP = {};
  v8f accS = {};

  for (int kc = 0; kc < INF_; kc += 32) {
    v16h a, bp, bs;
#pragma unroll
    for (int e = 0; e < 16; ++e) {
      int i = e >> 1;
      // 16-bit A 16x32 layout: VGPR i<4 -> K = 8*half + 2i (+pbit); i>=4 -> K = 16 + 8*half + 2(i-4)
      int k  = ((i & 4) << 2) + (h2 << 3) + ((i & 3) << 1) + (e & 1);
      int kg = kc + k;
      a[e]  = xt[id][kg];
      bp[e] = (_Float16)Wp[(size_t)kg * HF + wid * 16 + id];
      bs[e] = (_Float16)Ws[(size_t)kg * HF + wid * 16 + id];
    }
    accP = __builtin_amdgcn_wmma_f32_16x16x32_f16(false, a, false, bp, (short)0, accP, false, false);
    accS = __builtin_amdgcn_wmma_f32_16x16x32_f16(false, a, false, bs, (short)0, accS, false, false);
  }

  const float as = a_src[wid * NF + id];
  const float at = a_tgt[wid * NF + id];

#pragma unroll
  for (int r = 0; r < 8; ++r) {
    int m    = (h2 << 3) + r;            // C layout: VGPR r -> M = r + 8*half, N = lane%16
    int grow = row0 + m;
    size_t o = (size_t)grow * HF + wid * 16 + id;
    p[o]   = accP[r];
    out[o] = accS[r];                    // seed out with skip branch (pre-activation)

    float vs = accP[r] * as;
    float vt = accP[r] * at;
#pragma unroll
    for (int msk = 1; msk < 16; msk <<= 1) {
      vs += __shfl_xor(vs, msk, 32);
      vt += __shfl_xor(vt, msk, 32);
    }
    if (id == 0) {
      ssrc[grow * NH + wid] = vs;
      stgt[grow * NH + wid] = vt;
    }
  }
}

__global__ __launch_bounds__(256)
void gat_edge_max_kernel(const long long* __restrict__ ei,
                         const float* __restrict__ ssrc,
                         const float* __restrict__ stgt,
                         float* __restrict__ smax) {
  __shared__ float red[256];
  const int total = NE * NH;
  float m = -3.0e38f;
  for (int idx = blockIdx.x * blockDim.x + threadIdx.x; idx < total;
       idx += gridDim.x * blockDim.x) {
    int e = idx >> 3, h = idx & 7;
    int s = (int)ei[e];
    int t = (int)ei[NE + e];
    float v = ssrc[s * NH + h] + stgt[t * NH + h];
    v = (v > 0.0f) ? v : NEG_SLOPE * v;
    m = fmaxf(m, v);
  }
  red[threadIdx.x] = m;
  __syncthreads();
  for (int off = 128; off > 0; off >>= 1) {
    if (threadIdx.x < off) red[threadIdx.x] = fmaxf(red[threadIdx.x], red[threadIdx.x + off]);
    __syncthreads();
  }
  if (threadIdx.x == 0) atomicMaxF(smax, red[0]);
}

__global__ __launch_bounds__(256)
void gat_edge_denom_kernel(const long long* __restrict__ ei,
                           const float* __restrict__ ssrc,
                           const float* __restrict__ stgt,
                           const float* __restrict__ smax,
                           float* __restrict__ denom) {
  const int total = NE * NH;
  const float mx = *smax;
  for (int idx = blockIdx.x * blockDim.x + threadIdx.x; idx < total;
       idx += gridDim.x * blockDim.x) {
    int e = idx >> 3, h = idx & 7;
    int s = (int)ei[e];
    int t = (int)ei[NE + e];
    float v = ssrc[s * NH + h] + stgt[t * NH + h];
    v = (v > 0.0f) ? v : NEG_SLOPE * v;
    atomicAdd(&denom[t * NH + h], __expf(v - mx));
  }
}

__global__ __launch_bounds__(256)
void gat_edge_msg_kernel(const long long* __restrict__ ei,
                         const float* __restrict__ ssrc,
                         const float* __restrict__ stgt,
                         const float* __restrict__ smax,
                         const float* __restrict__ denom,
                         const float* __restrict__ p,
                         float* __restrict__ out) {
  const int total = NE * NH;
  const float mx = *smax;
  for (int idx = blockIdx.x * blockDim.x + threadIdx.x; idx < total;
       idx += gridDim.x * blockDim.x) {
    int e = idx >> 3, h = idx & 7;
    int s = (int)ei[e];
    int t = (int)ei[NE + e];
    float v = ssrc[s * NH + h] + stgt[t * NH + h];
    v = (v > 0.0f) ? v : NEG_SLOPE * v;
    float att = __expf(v - mx) / (denom[t * NH + h] + EPSV);

    const float4* prow = (const float4*)(p + (size_t)s * HF + h * NF);
    float* orow = out + (size_t)t * HF + h * NF;
#pragma unroll
    for (int q = 0; q < 4; ++q) {
      float4 pv = prow[q];
      atomicAdd(orow + q * 4 + 0, pv.x * att);
      atomicAdd(orow + q * 4 + 1, pv.y * att);
      atomicAdd(orow + q * 4 + 2, pv.z * att);
      atomicAdd(orow + q * 4 + 3, pv.w * att);
    }
  }
}

__global__ __launch_bounds__(256)
void gat_elu_kernel(float* __restrict__ out) {
  int i = blockIdx.x * blockDim.x + threadIdx.x;
  if (i < NN * HF) {
    float v = out[i];
    out[i] = (v > 0.0f) ? v : (__expf(v) - 1.0f);
  }
}

extern "C" void kernel_launch(void* const* d_in, const int* in_sizes, int n_in,
                              void* d_out, int out_size, void* d_ws, size_t ws_size,
                              hipStream_t stream) {
  const float*     x     = (const float*)d_in[0];
  const long long* ei    = (const long long*)d_in[1];  // int64 edge_index [2, NE]
  const float*     Wp    = (const float*)d_in[2];
  const float*     Wsk   = (const float*)d_in[3];
  const float*     a_src = (const float*)d_in[4];
  const float*     a_tgt = (const float*)d_in[5];
  float*           out   = (float*)d_out;

  // workspace layout (floats): p[NN*HF] | ssrc[NN*NH] | stgt[NN*NH] | denom[NN*NH] | smax[1]
  float* ws    = (float*)d_ws;
  float* p     = ws;
  float* ssrc  = p + (size_t)NN * HF;
  float* stgt  = ssrc + (size_t)NN * NH;
  float* denom = stgt + (size_t)NN * NH;
  float* smax  = denom + (size_t)NN * NH;

  gat_init_kernel<<<(NN * NH + 255) / 256, 256, 0, stream>>>(denom, smax);
  gat_gemm_scores_kernel<<<NN / 16, 256, 0, stream>>>(x, Wp, Wsk, a_src, a_tgt,
                                                      p, ssrc, stgt, out);
  gat_edge_max_kernel<<<2048, 256, 0, stream>>>(ei, ssrc, stgt, smax);
  gat_edge_denom_kernel<<<4096, 256, 0, stream>>>(ei, ssrc, stgt, smax, denom);
  gat_edge_msg_kernel<<<8192, 256, 0, stream>>>(ei, ssrc, stgt, smax, denom, p, out);
  gat_elu_kernel<<<(NN * HF + 255) / 256, 256, 0, stream>>>(out);
}
